// PulsarSceneModel_29154238005930
// MI455X (gfx1250) — matrix operand order repaired and checked by reading, not compiled, not run on hardware
//
#include <hip/hip_runtime.h>
#include <math.h>

// ---------------- problem constants (from reference) ----------------
#define HH     300
#define WW     400
#define HWPIX  (HH*WW)
#define NPTS   200000
#define DFEAT  256
#define HIDN   256
#define BATCH  2

#define TILE_M   64      // rows of points per workgroup (4 waves x 16)
#define THREADS  128     // 4 wave32
#define FPAD     264     // feat LDS row stride (ushorts) -> 528B rows, 16B aligned, bank-skewed
#define HPAD     40      // h-tile LDS row stride (ushorts) -> 80B rows, 16B aligned

typedef __attribute__((ext_vector_type(16))) __bf16 v16bf;
typedef __attribute__((ext_vector_type(8)))  float  v8f;

union Frag { v16bf v; uint4 q[2]; unsigned short s[16]; };
union Acc  { v8f v; float f[8]; };

__device__ __forceinline__ unsigned short f2bf(float f) {
    unsigned int u = __float_as_uint(f);
    u += 0x7FFFu + ((u >> 16) & 1u);     // round-to-nearest-even
    return (unsigned short)(u >> 16);
}

// ---------------- prep: weight repack (bf16) + accumulator zero ----------------
// W1T[n][k]   = bf16(W1[k][n])   (WMMA-B-fragment friendly: 32B contiguous per lane)
// W2pad[n][k] = n<3 ? bf16(W2[k][n]) : 0   (3 cols zero-padded to 16)
__global__ void prep_kernel(const float* __restrict__ W1, const float* __restrict__ W2,
                            unsigned short* __restrict__ W1T, unsigned short* __restrict__ W2pad,
                            float* __restrict__ accum) {
    int i = blockIdx.x * blockDim.x + threadIdx.x;
    if (i < 256*256) {
        int n = i >> 8, k = i & 255;
        W1T[n*256 + k] = f2bf(W1[k*256 + n]);
    } else if (i < 256*256 + 16*256) {
        int j = i - 256*256;
        int n = j >> 8, k = j & 255;
        W2pad[n*256 + k] = (n < 3) ? f2bf(W2[k*3 + n]) : (unsigned short)0;
    }
    // zero the splat accumulator every call (graph-replay safe)
    for (int j = i; j < BATCH*HWPIX*4; j += gridDim.x * blockDim.x) accum[j] = 0.f;
}

// ---------------- fused: geometry + GEMM1(+vdir rank-3) + relu + GEMM2 + sigmoid + splat ----------------
__global__ __launch_bounds__(THREADS)
void fused_kernel(const float* __restrict__ vert_pos, const float* __restrict__ pose,
                  const float* __restrict__ intr,     const float* __restrict__ feat,
                  const float* __restrict__ opy,      const float* __restrict__ W1,
                  const float* __restrict__ b1,       const float* __restrict__ b2,
                  const unsigned short* __restrict__ W1T,
                  const unsigned short* __restrict__ W2pad,
                  float* __restrict__ accum) {
    __shared__ alignas(16) unsigned short featS[TILE_M * FPAD];   // 33.8 KB bf16 feat tile
    __shared__ alignas(16) unsigned short hS[4][16 * HPAD];       // per-wave h transpose buffer
    __shared__ float vdirS[BATCH][TILE_M][3];
    __shared__ int   idxS[BATCH][TILE_M];
    __shared__ float wS[BATCH][TILE_M];
    __shared__ float rgbS[4][16][4];

    const int tid  = threadIdx.x;
    const int base = blockIdx.x * TILE_M;

    // ---- stage 0a: stage feat tile -> bf16 in LDS (single pass over the 205MB array) ----
    for (int i = tid; i < TILE_M * 64; i += THREADS) {
        int row = i >> 6, c4 = (i & 63) << 2;
        int rg = base + row;
        float4 f = make_float4(0.f, 0.f, 0.f, 0.f);
        if (rg < NPTS) f = *(const float4*)(feat + (size_t)rg * DFEAT + c4);
        unsigned int p0 = (unsigned int)f2bf(f.x) | ((unsigned int)f2bf(f.y) << 16);
        unsigned int p1 = (unsigned int)f2bf(f.z) | ((unsigned int)f2bf(f.w) << 16);
        *(uint2*)&featS[row * FPAD + c4] = make_uint2(p0, p1);
    }

    // ---- stage 0b: per-point geometry (project, validity, view dir) ----
    if (tid < TILE_M) {
        int rg = base + tid;
        bool inr = rg < NPTS;
        float x = 0.f, y = 0.f, z = 1.f;
        if (inr) { x = vert_pos[rg]; y = vert_pos[NPTS + rg]; z = vert_pos[2*NPTS + rg]; }
        float op = 1.f / (1.f + __expf(-(inr ? opy[rg] : 0.f)));   // sigmoid(opacity)
        for (int b = 0; b < BATCH; ++b) {
            const float* P  = pose + b * 16;
            const float* Km = intr + b * 9;
            float xt = P[0]*x + P[1]*y + P[2]*z  + P[3];
            float yt = P[4]*x + P[5]*y + P[6]*z  + P[7];
            float zt = P[8]*x + P[9]*y + P[10]*z + P[11];
            float px = Km[0]*xt + Km[1]*yt + Km[2]*zt;
            float py = Km[3]*xt + Km[4]*yt + Km[5]*zt;
            float pz = Km[6]*xt + Km[7]*yt + Km[8]*zt;
            bool  epsm = fabsf(pz) < 0.01f;
            float zsv  = epsm ? 0.01f : pz;
            float uu   = epsm ? -1.0e6f : px / zsv;
            float vv   = epsm ? -1.0e6f : py / zsv;
            float zz   = epsm ? -1.0e6f : zsv;
            int ix = (int)rintf(uu), iy = (int)rintf(vv);       // round-half-even matches jnp.round
            bool valid = inr && (ix >= 0) && (ix < WW) && (iy >= 0) && (iy < HH) && (zz >= 1.0f);
            idxS[b][tid] = valid ? (iy * WW + ix) : 0;
            wS[b][tid]   = valid ? op : 0.f;
            // vdir = inv(R) @ normalize(xyz_t)   (3x3 adjugate inverse of pose rotation block)
            float rn = rsqrtf(xt*xt + yt*yt + zt*zt);
            float nx = xt*rn, ny = yt*rn, nz = zt*rn;
            float a00=P[0],a01=P[1],a02=P[2],a10=P[4],a11=P[5],a12=P[6],a20=P[8],a21=P[9],a22=P[10];
            float C00=a11*a22-a12*a21, C01=a12*a20-a10*a22, C02=a10*a21-a11*a20;
            float C10=a02*a21-a01*a22, C11=a00*a22-a02*a20, C12=a01*a20-a00*a21;
            float C20=a01*a12-a02*a11, C21=a02*a10-a00*a12, C22=a00*a11-a01*a10;
            float id = 1.f / (a00*C00 + a01*C10 + a02*C20);
            vdirS[b][tid][0] = (C00*nx + C10*ny + C20*nz) * id;
            vdirS[b][tid][1] = (C01*nx + C11*ny + C21*nz) * id;
            vdirS[b][tid][2] = (C02*nx + C12*ny + C22*nz) * id;
        }
    }
    __syncthreads();

    // ---- main: per-wave 16-row tile, WMMA bf16 pipeline ----
    const int wave = tid >> 5;
    const int lid  = tid & 31;
    const int lo   = lid & 15;
    const int hi   = lid >> 4;
    const int wrow0 = wave * 16;

    // rows covered by this lane's C-layout elements: M = v + 8*hi
    float vd[BATCH][8][3];
    for (int b = 0; b < BATCH; ++b)
        #pragma unroll
        for (int v = 0; v < 8; ++v) {
            int rl = wrow0 + v + 8*hi;
            vd[b][v][0] = vdirS[b][rl][0];
            vd[b][v][1] = vdirS[b][rl][1];
            vd[b][v][2] = vdirS[b][rl][2];
        }

    Acc c2[BATCH];
    for (int b = 0; b < BATCH; ++b)
        for (int e = 0; e < 8; ++e) c2[b].f[e] = 0.f;

    for (int nc = 0; nc < HIDN; nc += 32) {
        float biasT[2], w1vT[2][3];
        #pragma unroll
        for (int t = 0; t < 2; ++t) {
            int n = nc + t*16 + lo;
            biasT[t]   = b1[n];
            w1vT[t][0] = W1[(256 + 0)*HIDN + n];
            w1vT[t][1] = W1[(256 + 1)*HIDN + n];
            w1vT[t][2] = W1[(256 + 2)*HIDN + n];
        }
        Acc cT[2];
        for (int t = 0; t < 2; ++t) for (int e = 0; e < 8; ++e) cT[t].f[e] = 0.f;

        const unsigned short* arow = &featS[(wrow0 + lo) * FPAD];
        #pragma unroll
        for (int kk = 0; kk < DFEAT; kk += 32) {
            Frag a;
            a.q[0] = *(const uint4*)(arow + kk + hi*16);       // K = hi*16 .. +7
            a.q[1] = *(const uint4*)(arow + kk + hi*16 + 8);   // K = hi*16+8 .. +7
            Frag bf0;
            const unsigned short* bp0 = W1T + (size_t)(nc + lo) * 256 + kk + hi*16;
            bf0.q[0] = *(const uint4*)(bp0);
            bf0.q[1] = *(const uint4*)(bp0 + 8);
            cT[0].v = __builtin_amdgcn_wmma_f32_16x16x32_bf16(false, a.v, false, bf0.v,
                                                              (short)0, cT[0].v, false, false);
            Frag bf1;
            const unsigned short* bp1 = W1T + (size_t)(nc + 16 + lo) * 256 + kk + hi*16;
            bf1.q[0] = *(const uint4*)(bp1);
            bf1.q[1] = *(const uint4*)(bp1 + 8);
            cT[1].v = __builtin_amdgcn_wmma_f32_16x16x32_bf16(false, a.v, false, bf1.v,
                                                              (short)0, cT[1].v, false, false);
        }

        // per-batch: rank-3 vdir correction + bias + relu, transpose via LDS, feed GEMM2 (K=32 step)
        unsigned short* hrow = &hS[wave][0];
        for (int b = 0; b < BATCH; ++b) {
            #pragma unroll
            for (int t = 0; t < 2; ++t)
                #pragma unroll
                for (int v = 0; v < 8; ++v) {
                    float val = cT[t].f[v] + biasT[t]
                              + w1vT[t][0]*vd[b][v][0] + w1vT[t][1]*vd[b][v][1] + w1vT[t][2]*vd[b][v][2];
                    val = fmaxf(val, 0.f);
                    hrow[(v + 8*hi) * HPAD + t*16 + lo] = f2bf(val);   // (row M, k=col-in-32)
                }
            // wave-local LDS transpose read as A fragment (DS ops are in-order per wave)
            Frag a2;
            const unsigned short* h0 = &hS[wave][lo * HPAD];
            a2.q[0] = *(const uint4*)(h0 + hi*16);
            a2.q[1] = *(const uint4*)(h0 + hi*16 + 8);
            Frag b2f;
            const unsigned short* wp = W2pad + (size_t)lo * 256 + nc + hi*16;
            b2f.q[0] = *(const uint4*)(wp);
            b2f.q[1] = *(const uint4*)(wp + 8);
            c2[b].v = __builtin_amdgcn_wmma_f32_16x16x32_bf16(false, a2.v, false, b2f.v,
                                                              (short)0, c2[b].v, false, false);
        }
    }

    // ---- epilogue: sigmoid + atomic splat ----
    for (int b = 0; b < BATCH; ++b) {
        if (lo < 3) {
            float bb = b2[lo];
            #pragma unroll
            for (int v = 0; v < 8; ++v) {
                float val = c2[b].f[v] + bb;
                rgbS[wave][v + 8*hi][lo] = 1.f / (1.f + __expf(-val));
            }
        }
        if (hi == 0) {                  // one lane per row
            int rl = wrow0 + lo;
            float w = wS[b][rl];
            if (w > 0.f) {
                int p = idxS[b][rl];
                float* acc = accum + ((size_t)b * HWPIX + p) * 4;
                unsafeAtomicAdd(acc + 0, w * rgbS[wave][lo][0]);
                unsafeAtomicAdd(acc + 1, w * rgbS[wave][lo][1]);
                unsafeAtomicAdd(acc + 2, w * rgbS[wave][lo][2]);
                unsafeAtomicAdd(acc + 3, w);
            }
        }
    }
}

// ---------------- compose: fg/alpha blend -> (B,3,H,W) ----------------
__global__ void compose_kernel(const float* __restrict__ accum, const float* __restrict__ bkg,
                               float* __restrict__ out) {
    int i = blockIdx.x * blockDim.x + threadIdx.x;
    if (i >= BATCH * HWPIX) return;
    int b = i / HWPIX, p = i % HWPIX;
    const float* a = accum + (size_t)i * 4;
    float den   = a[3];
    float alpha = 1.f - __expf(-den);
    float inv   = 1.f / (den + 1e-8f);
    #pragma unroll
    for (int c = 0; c < 3; ++c) {
        float fg = a[c] * inv;
        out[((size_t)(b * 3 + c)) * HWPIX + p] = fg * alpha + bkg[c] * (1.f - alpha);
    }
}

extern "C" void kernel_launch(void* const* d_in, const int* in_sizes, int n_in,
                              void* d_out, int out_size, void* d_ws, size_t ws_size,
                              hipStream_t stream) {
    const float* vert_pos = (const float*)d_in[0];
    // d_in[1] = ref_images (unused by the reference computation)
    const float* pose = (const float*)d_in[2];
    const float* intr = (const float*)d_in[3];
    const float* feat = (const float*)d_in[4];
    const float* opy  = (const float*)d_in[5];
    const float* W1   = (const float*)d_in[6];
    const float* b1   = (const float*)d_in[7];
    const float* W2   = (const float*)d_in[8];
    const float* b2   = (const float*)d_in[9];
    const float* bkg  = (const float*)d_in[10];
    float* out = (float*)d_out;

    char* ws = (char*)d_ws;
    unsigned short* W1T   = (unsigned short*)ws;                       // 131072 B
    unsigned short* W2pad = (unsigned short*)(ws + 131072);            // 8192 B
    float*          accum = (float*)(ws + 131072 + 8192);              // 3,840,000 B

    {
        int total = 256*256 + 16*256;
        int blk = (total + 255) / 256;
        prep_kernel<<<blk, 256, 0, stream>>>(W1, W2, W1T, W2pad, accum);
    }
    {
        int blk = (NPTS + TILE_M - 1) / TILE_M;
        fused_kernel<<<blk, THREADS, 0, stream>>>(vert_pos, pose, intr, feat, opy,
                                                  W1, b1, b2, W1T, W2pad, accum);
    }
    {
        int total = BATCH * HWPIX;
        int blk = (total + 255) / 256;
        compose_kernel<<<blk, 256, 0, stream>>>(accum, bkg, out);
    }
}
